// BudgetedSparseMAPPOAgent_88648124989767
// MI455X (gfx1250) — matrix operand order repaired and checked by reading, not compiled, not run on hardware
//
#include <hip/hip_runtime.h>
#include <hip/hip_bf16.h>
#include <math.h>

// ---------------------------------------------------------------------------
// CDNA5 WMMA bf16 helpers (wave32, 16x16x32, f32 accumulate)
// ---------------------------------------------------------------------------
typedef __attribute__((ext_vector_type(16))) __bf16 bf16x16;
typedef __attribute__((ext_vector_type(8)))  float  f32x8;

union BfVec { bf16x16 v; __bf16 e[16]; };

__device__ __forceinline__ f32x8 zero8() {
  f32x8 z = {0.f, 0.f, 0.f, 0.f, 0.f, 0.f, 0.f, 0.f};
  return z;
}

__device__ __forceinline__ float sigm(float x) { return 1.f / (1.f + expf(-x)); }

// A fragment (16x32 MxK, bf16): lane&15 = M row; lane[4] selects K-half.
// Elements 0..7 -> K = koff+0..7, elements 8..15 -> K = 16+koff+0..7.
__device__ __forceinline__ BfVec load_a_bf(const __bf16* __restrict__ src, int ld, int lane) {
  const int m = lane & 15;
  const int koff = (lane >> 4) << 3;
  const __bf16* p = src + m * ld + koff;
  BfVec a;
#pragma unroll
  for (int i = 0; i < 8; ++i) a.e[i] = p[i];
#pragma unroll
  for (int i = 0; i < 8; ++i) a.e[8 + i] = p[16 + i];
  return a;
}

// B fragment (32x16 KxN) sourced from a row-major weight W[N][K]:
// B[k][n] = W[n][k]; lane&15 = N, lane[4] selects K-half (same order as A):
// elements 0..7 -> K = koff+0..7, elements 8..15 -> K = 16+koff+0..7.
__device__ __forceinline__ BfVec load_b_bf(const __bf16* __restrict__ w, int ld, int lane) {
  const int n = lane & 15;
  const int kb = (lane >> 4) << 3;
  const __bf16* p = w + n * ld + kb;
  BfVec b;
#pragma unroll
  for (int i = 0; i < 8; ++i) b.e[i] = p[i];
#pragma unroll
  for (int i = 0; i < 8; ++i) b.e[8 + i] = p[16 + i];
  return b;
}

#define WMMA_BF16(a, b, c) \
  __builtin_amdgcn_wmma_f32_16x16x32_bf16(false, (a), false, (b), (short)0, (c), false, false)

// ---------------------------------------------------------------------------
// Problem constants
// ---------------------------------------------------------------------------
#define NROWS   65536      // BS * A
#define NAG     32
#define HDIM    16
#define CVDIM   16
#define NACT    20
#define PQKVG   208        // 64 q + 64 k + 64 v + 4 g, padded to 13*16
#define PIN     320        // HID + NH*CV

// ws layout (element offsets in __bf16 units unless noted)
#define WS_HBF     0u
#define WS_FC1W    16777216u                 // + 65536
#define WS_WIH     (WS_FC1W + 65536u)        // + 196608
#define WS_WHH     (WS_WIH + 196608u)        // + 196608
#define WS_WQKVG   (WS_WHH + 196608u)        // + 53248
#define WS_P1W     (WS_WQKVG + 53248u)       // + 81920
#define WS_P2W     (WS_P1W + 81920u)         // + 8192
#define WS_END_U16 (WS_P2W + 8192u)          // f32 qkvg bias starts here (byte /4 aligned)

// ---------------------------------------------------------------------------
// Kernel 0: weight conversion / packing (f32 -> bf16) into workspace
// ---------------------------------------------------------------------------
__global__ __launch_bounds__(256) void k0_prep(
    const float* __restrict__ fc1_w, const float* __restrict__ wih, const float* __restrict__ whh,
    const float* __restrict__ qw, const float* __restrict__ qb,
    const float* __restrict__ kw, const float* __restrict__ kb,
    const float* __restrict__ vw, const float* __restrict__ vb,
    const float* __restrict__ gw, const float* __restrict__ gb,
    const float* __restrict__ p1w, const float* __restrict__ p2w,
    __bf16* __restrict__ o_fc1, __bf16* __restrict__ o_wih,
    __bf16* __restrict__ o_whh, __bf16* __restrict__ o_qkvg,
    float* __restrict__ o_bqkvg, __bf16* __restrict__ o_p1,
    __bf16* __restrict__ o_p2)
{
  const int t0 = blockIdx.x * blockDim.x + threadIdx.x;
  const int st = gridDim.x * blockDim.x;
  for (int i = t0; i < 256 * 256; i += st) o_fc1[i] = (__bf16)fc1_w[i];
  for (int i = t0; i < 768 * 256; i += st) o_wih[i] = (__bf16)wih[i];
  for (int i = t0; i < 768 * 256; i += st) o_whh[i] = (__bf16)whh[i];
  for (int i = t0; i < PQKVG * 256; i += st) {
    const int r = i >> 8, c = i & 255;
    float v = 0.f;
    if      (r < 64)  v = qw[r * 256 + c];
    else if (r < 128) v = kw[(r - 64) * 256 + c];
    else if (r < 192) v = vw[(r - 128) * 256 + c];
    else if (r < 196) v = gw[(r - 192) * 256 + c];
    o_qkvg[i] = (__bf16)v;
  }
  for (int i = t0; i < PQKVG; i += st) {
    float v = 0.f;
    if      (i < 64)  v = qb[i];
    else if (i < 128) v = kb[i - 64];
    else if (i < 192) v = vb[i - 128];
    else if (i < 196) v = gb[i - 192];
    o_bqkvg[i] = v;
  }
  for (int i = t0; i < 256 * PIN; i += st) o_p1[i] = (__bf16)p1w[i];
  for (int i = t0; i < 32 * 256; i += st) {
    const int r = i >> 8;
    o_p2[i] = (r < NACT) ? (__bf16)p2w[i] : (__bf16)0.f;
  }
}

// ---------------------------------------------------------------------------
// Kernel 1: fc1 + GRU fused. 64 rows/block, 8 waves, 64KB LDS.
// Inputs are converted to bf16 ONCE per block (vectorized), then all WMMA
// A-fragments are plain LDS reads.
// ---------------------------------------------------------------------------
__global__ __launch_bounds__(256) void k1_fc1_gru(
    const float* __restrict__ in, const float* __restrict__ hin,
    const float* __restrict__ fc1_b,
    const float* __restrict__ b_ih, const float* __restrict__ b_hh,
    const __bf16* __restrict__ fc1w,
    const __bf16* __restrict__ wih,
    const __bf16* __restrict__ whh,
    float* __restrict__ out_h,
    __bf16* __restrict__ h_bf)
{
  __shared__ __bf16 sX[64 * 256];    // relu(fc1) activations         (32KB)
  __shared__ __bf16 sA[64 * 256];    // phase A: in_bf / phase B: h_bf (32KB)
  const int tid = threadIdx.x;
  const int lane = tid & 31;
  const int wave = tid >> 5;
  const int row0 = blockIdx.x * 64;
  const int n = lane & 15;
  const int hi = (lane >> 4) << 3;   // D-layout: M = r + hi

  // Stage 0: convert input tile (64x256 f32) -> bf16 LDS, vectorized.
  {
    const float4* in4 = (const float4*)(in + row0 * 256);
    for (int i = tid; i < 64 * 256 / 4; i += 256) {
      const float4 t = in4[i];
      __bf16* d = &sA[i * 4];
      d[0] = (__bf16)t.x; d[1] = (__bf16)t.y; d[2] = (__bf16)t.z; d[3] = (__bf16)t.w;
    }
  }
  __syncthreads();

  // Phase A: x = relu(in @ fc1_w^T + fc1_b) -> LDS bf16
  for (int task = wave; task < 64; task += 8) {
    const int rt = task >> 4, ct = task & 15;
    f32x8 acc = zero8();
#pragma unroll
    for (int ks = 0; ks < 8; ++ks) {
      const int k0 = ks * 32;
      BfVec a = load_a_bf(sA + (rt * 16) * 256 + k0, 256, lane);
      BfVec b = load_b_bf(fc1w + (ct * 16) * 256 + k0, 256, lane);
      acc = WMMA_BF16(a.v, b.v, acc);
    }
    const float bias = fc1_b[ct * 16 + n];
#pragma unroll
    for (int r = 0; r < 8; ++r) {
      const float v = acc[r] + bias;
      sX[(rt * 16 + r + hi) * 256 + ct * 16 + n] = (__bf16)(v > 0.f ? v : 0.f);
    }
  }
  __syncthreads();

  // Stage 0b: overwrite sA with the hidden-state tile (bf16).
  {
    const float4* h4 = (const float4*)(hin + row0 * 256);
    for (int i = tid; i < 64 * 256 / 4; i += 256) {
      const float4 t = h4[i];
      __bf16* d = &sA[i * 4];
      d[0] = (__bf16)t.x; d[1] = (__bf16)t.y; d[2] = (__bf16)t.z; d[3] = (__bf16)t.w;
    }
  }
  __syncthreads();

  // Phase B: gi = x@Wih^T, gh = h@Whh^T; six accumulator tiles per task,
  // then GRU gate math directly on the WMMA D-layout.
  for (int task = wave; task < 64; task += 8) {
    const int rt = task >> 4, ct = task & 15;
    f32x8 ir = zero8(), iz = zero8(), inn = zero8();
    f32x8 hr = zero8(), hz = zero8(), hnn = zero8();
#pragma unroll 2
    for (int ks = 0; ks < 8; ++ks) {
      const int k0 = ks * 32;
      BfVec ax = load_a_bf(sX + (rt * 16) * 256 + k0, 256, lane);
      BfVec ah = load_a_bf(sA + (rt * 16) * 256 + k0, 256, lane);
      BfVec b;
      b = load_b_bf(wih + (      ct * 16) * 256 + k0, 256, lane); ir  = WMMA_BF16(ax.v, b.v, ir);
      b = load_b_bf(wih + (256 + ct * 16) * 256 + k0, 256, lane); iz  = WMMA_BF16(ax.v, b.v, iz);
      b = load_b_bf(wih + (512 + ct * 16) * 256 + k0, 256, lane); inn = WMMA_BF16(ax.v, b.v, inn);
      b = load_b_bf(whh + (      ct * 16) * 256 + k0, 256, lane); hr  = WMMA_BF16(ah.v, b.v, hr);
      b = load_b_bf(whh + (256 + ct * 16) * 256 + k0, 256, lane); hz  = WMMA_BF16(ah.v, b.v, hz);
      b = load_b_bf(whh + (512 + ct * 16) * 256 + k0, 256, lane); hnn = WMMA_BF16(ah.v, b.v, hnn);
    }
    const int c = ct * 16 + n;
    const float bir = b_ih[c], biz = b_ih[256 + c], bin = b_ih[512 + c];
    const float bhr = b_hh[c], bhz = b_hh[256 + c], bhn = b_hh[512 + c];
#pragma unroll
    for (int r = 0; r < 8; ++r) {
      const int grow = row0 + rt * 16 + r + hi;
      const float rg = sigm((ir[r] + bir) + (hr[r] + bhr));
      const float zg = sigm((iz[r] + biz) + (hz[r] + bhz));
      const float ng = tanhf((inn[r] + bin) + rg * (hnn[r] + bhn));
      const float hp = hin[grow * 256 + c];
      const float h  = (1.f - zg) * ng + zg * hp;
      out_h[grow * 256 + c] = h;
      h_bf[grow * 256 + c]  = (__bf16)h;
    }
  }
}

// ---------------------------------------------------------------------------
// Kernel 2: qkvg projection + sparse top-8 attention + policy head.
// 1 batch (32 agents)/block, 8 waves, 62KB LDS.
// ---------------------------------------------------------------------------
__global__ __launch_bounds__(256) void k2_attn_policy(
    const __bf16* __restrict__ h_bf,
    const __bf16* __restrict__ wqkvg, const float* __restrict__ bqkvg,
    const __bf16* __restrict__ p1w, const float* __restrict__ p1_b,
    const __bf16* __restrict__ p2w, const float* __restrict__ p2_b,
    float* __restrict__ out_logits)
{
  __shared__ float sQ[NAG * PQKVG];        // packed q|k|v|g, f32: 26.6KB
  __shared__ __bf16 sPol[NAG * PIN];       // [h, gated] bf16: 20KB
  union Scr { float sc[128 * 32]; __bf16 hid[NAG * 256]; };
  __shared__ Scr sS;                       // attention scores / p1 output: 16KB

  const int tid = threadIdx.x, lane = tid & 31, wave = tid >> 5;
  const int row0 = blockIdx.x * NAG;
  const int n = lane & 15, hi = (lane >> 4) << 3;

  // Phase 1: packed [q k v g] = h @ Wqkvg^T + b -> f32 LDS
  for (int task = wave; task < 26; task += 8) {
    const int rt = task / 13, ct = task % 13;
    f32x8 acc = zero8();
#pragma unroll
    for (int ks = 0; ks < 8; ++ks) {
      const int k0 = ks * 32;
      BfVec a = load_a_bf(h_bf + (row0 + rt * 16) * 256 + k0, 256, lane);
      BfVec b = load_b_bf(wqkvg + (ct * 16) * 256 + k0, 256, lane);
      acc = WMMA_BF16(a.v, b.v, acc);
    }
    const float bias = bqkvg[ct * 16 + n];
#pragma unroll
    for (int r = 0; r < 8; ++r)
      sQ[(rt * 16 + r + hi) * PQKVG + ct * 16 + n] = acc[r] + bias;
  }
  // Stage h (already bf16) into the 320-wide policy input
  for (int i = tid; i < NAG * 256; i += 256)
    sPol[(i >> 8) * PIN + (i & 255)] = h_bf[(row0 + (i >> 8)) * 256 + (i & 255)];
  __syncthreads();

  // Phase 2: per-(agent q, head hh) scores, exact top-8, softmax, gated msg
  if (tid < 128) {
    const int q = tid >> 2, hh = tid & 3;
    float* sc = &sS.sc[tid * 32];
    const float* qv = &sQ[q * PQKVG + hh * HDIM];
    for (int k = 0; k < NAG; ++k) {
      const float* kv = &sQ[k * PQKVG + 64 + hh * HDIM];
      float s = 0.f;
#pragma unroll
      for (int j = 0; j < HDIM; ++j) s += qv[j] * kv[j];
      sc[k] = s * 0.25f;                 // 1/sqrt(16)
    }
    sc[q] = -1e10f;                      // mask self
    unsigned kept = 0u;                  // top-8, ties -> smallest index (lax.top_k)
    for (int t = 0; t < 8; ++t) {
      float best = -3.0e38f; int bi = 0;
      for (int k = 0; k < NAG; ++k)
        if (!((kept >> k) & 1u) && sc[k] > best) { best = sc[k]; bi = k; }
      kept |= 1u << bi;
    }
    float smax = -3.0e38f;
    for (int k = 0; k < NAG; ++k) if ((kept >> k) & 1u) smax = fmaxf(smax, sc[k]);
    float den = 0.f;
    float msg[CVDIM];
#pragma unroll
    for (int j = 0; j < CVDIM; ++j) msg[j] = 0.f;
    for (int k = 0; k < NAG; ++k) {
      if (!((kept >> k) & 1u)) continue;
      const float w = expf(sc[k] - smax);
      den += w;
      const float* vv = &sQ[k * PQKVG + 128 + hh * CVDIM];
#pragma unroll
      for (int j = 0; j < CVDIM; ++j) msg[j] += w * vv[j];
    }
    const float gate = sigm(sQ[q * PQKVG + 192 + hh]);
    const float gs = gate / den;
    __bf16* dst = &sPol[q * PIN + 256 + hh * CVDIM];
#pragma unroll
    for (int j = 0; j < CVDIM; ++j) dst[j] = (__bf16)(msg[j] * gs);
  }
  __syncthreads();

  // Phase 3: p1 = relu(policy_in @ p1_w^T + b), K = 320
  for (int task = wave; task < 32; task += 8) {
    const int rt = task >> 4, ct = task & 15;
    f32x8 acc = zero8();
#pragma unroll
    for (int ks = 0; ks < 10; ++ks) {
      const int k0 = ks * 32;
      BfVec a = load_a_bf(sPol + (rt * 16) * PIN + k0, PIN, lane);
      BfVec b = load_b_bf(p1w + (ct * 16) * PIN + k0, PIN, lane);
      acc = WMMA_BF16(a.v, b.v, acc);
    }
    const float bias = p1_b[ct * 16 + n];
#pragma unroll
    for (int r = 0; r < 8; ++r) {
      const float v = acc[r] + bias;
      sS.hid[(rt * 16 + r + hi) * 256 + ct * 16 + n] = (__bf16)(v > 0.f ? v : 0.f);
    }
  }
  __syncthreads();

  // Phase 4: logits = hid @ p2_w^T + b (N = 20, weight rows padded to 32)
  if (wave < 4) {
    const int rt = wave >> 1, ct = wave & 1;
    f32x8 acc = zero8();
#pragma unroll
    for (int ks = 0; ks < 8; ++ks) {
      const int k0 = ks * 32;
      BfVec a = load_a_bf(sS.hid + (rt * 16) * 256 + k0, 256, lane);
      BfVec b = load_b_bf(p2w + (ct * 16) * 256 + k0, 256, lane);
      acc = WMMA_BF16(a.v, b.v, acc);
    }
    const int nn = ct * 16 + n;
    if (nn < NACT) {
      const float bias = p2_b[nn];
#pragma unroll
      for (int r = 0; r < 8; ++r)
        out_logits[(row0 + rt * 16 + r + hi) * NACT + nn] = acc[r] + bias;
    }
  }
}

// ---------------------------------------------------------------------------
// Host launcher
// ---------------------------------------------------------------------------
extern "C" void kernel_launch(void* const* d_in, const int* in_sizes, int n_in,
                              void* d_out, int out_size, void* d_ws, size_t ws_size,
                              hipStream_t stream) {
  (void)in_sizes; (void)n_in; (void)out_size; (void)ws_size;
  const float* in    = (const float*)d_in[0];
  const float* hin   = (const float*)d_in[1];
  const float* fc1_w = (const float*)d_in[2];
  const float* fc1_b = (const float*)d_in[3];
  const float* wih   = (const float*)d_in[4];
  const float* whh   = (const float*)d_in[5];
  const float* b_ih  = (const float*)d_in[6];
  const float* b_hh  = (const float*)d_in[7];
  const float* q_w   = (const float*)d_in[8];
  const float* q_b   = (const float*)d_in[9];
  const float* k_w   = (const float*)d_in[10];
  const float* k_b   = (const float*)d_in[11];
  const float* v_w   = (const float*)d_in[12];
  const float* v_b   = (const float*)d_in[13];
  const float* g_w   = (const float*)d_in[14];
  const float* g_b   = (const float*)d_in[15];
  const float* p1_w  = (const float*)d_in[16];
  const float* p1_b  = (const float*)d_in[17];
  const float* p2_w  = (const float*)d_in[18];
  const float* p2_b  = (const float*)d_in[19];

  __bf16* ws_b = (__bf16*)d_ws;
  __bf16* h_bf  = ws_b + WS_HBF;
  __bf16* fc1wb = ws_b + WS_FC1W;
  __bf16* wihb  = ws_b + WS_WIH;
  __bf16* whhb  = ws_b + WS_WHH;
  __bf16* wqkvg = ws_b + WS_WQKVG;
  __bf16* p1wb  = ws_b + WS_P1W;
  __bf16* p2wb  = ws_b + WS_P2W;
  float*  bqkvg = (float*)(ws_b + WS_END_U16);

  float* out_logits = (float*)d_out;                 // [65536, 20]
  float* out_h      = out_logits + NROWS * NACT;     // [65536, 256]

  k0_prep<<<256, 256, 0, stream>>>(fc1_w, wih, whh, q_w, q_b, k_w, k_b,
                                   v_w, v_b, g_w, g_b, p1_w, p2_w,
                                   fc1wb, wihb, whhb, wqkvg, bqkvg, p1wb, p2wb);

  k1_fc1_gru<<<NROWS / 64, 256, 0, stream>>>(in, hin, fc1_b, b_ih, b_hh,
                                             fc1wb, wihb, whhb, out_h, h_bf);

  k2_attn_policy<<<NROWS / NAG, 256, 0, stream>>>(h_bf, wqkvg, bqkvg,
                                                  p1wb, p1_b, p2wb, p2_b,
                                                  out_logits);
}
